// NodeModel_13486197310233
// MI455X (gfx1250) — matrix operand order repaired and checked by reading, not compiled, not run on hardware
//
#include <hip/hip_runtime.h>
#include <hip/hip_bf16.h>

#define NN   40000
#define NE   640000
#define DN   128
#define HID  256
#define DOUT 128

typedef __attribute__((ext_vector_type(16))) __bf16       v16bf;
typedef __attribute__((ext_vector_type(8)))  float        v8f;
typedef __attribute__((ext_vector_type(4)))  unsigned int u32x4;

__device__ __forceinline__ unsigned short f32_to_bf16(float f) {
  unsigned u = __float_as_uint(f);
  u += 0x7FFFu + ((u >> 16) & 1u);   // round-to-nearest-even
  return (unsigned short)(u >> 16);
}

// Build a 16-element bf16 fragment from two 16-byte LDS chunks.
__device__ __forceinline__ v16bf load_frag2(const unsigned short* p0,
                                            const unsigned short* p1) {
  union { u32x4 q[2]; v16bf v; } u;
  u.q[0] = *(const u32x4*)p0;
  u.q[1] = *(const u32x4*)p1;
  return u.v;
}

// Fused GEMM layer: acc[M=128, N=NT*16] += hin[128,K] * W[K,N]
// hin: LDS bf16, row pitch inPad halves.
// Wt : global bf16, PRE-TRANSPOSED [N][K] row-major.
// wt0/wt1: double-buffered LDS panels, 256 rows x 48-half pitch.
template<int K, int NT>
__device__ __forceinline__ void mlp_layer(const unsigned short* hin, int inPad,
                                          const unsigned short* __restrict__ Wt,
                                          unsigned short* wt0, unsigned short* wt1,
                                          v8f* acc) {
  constexpr int N  = NT * 16;
  constexpr int CH = (N * 4) / 256;   // 16B chunks per thread per panel
  constexpr int KP = K / 32;
  const int tid  = threadIdx.x;
  const int lane = tid & 31;
  const int wv   = tid >> 5;
  const int arow = wv * 16 + (lane & 15);   // A: lanes hold rows (ISA 7.12.2)
  const int ksel = (lane >> 4);             // lane hi-half selects K sub-block

#pragma unroll
  for (int nt = 0; nt < NT; ++nt) {
    v8f z = {0.f, 0.f, 0.f, 0.f, 0.f, 0.f, 0.f, 0.f};
    acc[nt] = z;
  }

  // Stage panel 0 (k = 0..31): pure 16B vector copies, Wt rows are contiguous.
#pragma unroll
  for (int c = 0; c < CH; ++c) {
    const int chunk = tid + c * 256;
    const int n = chunk >> 2, q = chunk & 3;
    *(u32x4*)&wt0[n * 48 + q * 8] = *(const u32x4*)&Wt[n * K + q * 8];
  }
  __syncthreads();

  unsigned short* bufs[2] = {wt0, wt1};
  for (int kp = 0; kp < KP; ++kp) {
    unsigned short* cur = bufs[kp & 1];
    unsigned short* nxt = bufs[(kp + 1) & 1];

    // Issue global loads for the NEXT panel now; they fly during the wmmas.
    u32x4 pre[CH];
    const int k0n = (kp + 1) * 32;
    if (kp + 1 < KP) {
#pragma unroll
      for (int c = 0; c < CH; ++c) {
        const int chunk = tid + c * 256;
        const int n = chunk >> 2, q = chunk & 3;
        pre[c] = *(const u32x4*)&Wt[n * K + k0n + q * 8];
      }
    }

    // A fragment 16x32: lane<16 holds K 0-7 & 16-23; lane>=16 holds 8-15 & 24-31
    const unsigned short* ap = hin + arow * inPad + kp * 32 + ksel * 8;
    v16bf a = load_frag2(ap, ap + 16);

    // B fragments, software-pipelined so ds_loads overlap the matrix pipe.
    const unsigned short* bbase = cur + (lane & 15) * 48 + ksel * 8;
    v16bf b0 = load_frag2(bbase, bbase + 16);
#pragma unroll
    for (int nt = 0; nt < NT; ++nt) {
      v16bf bn = b0;
      if (nt + 1 < NT) {
        const unsigned short* bp = bbase + (nt + 1) * 16 * 48;
        bn = load_frag2(bp, bp + 16);
      }
      acc[nt] = __builtin_amdgcn_wmma_f32_16x16x32_bf16(
          false, a, false, b0, (short)0, acc[nt], false, false);
      b0 = bn;
    }

    // Commit the staged panel to the other LDS buffer.
    if (kp + 1 < KP) {
#pragma unroll
      for (int c = 0; c < CH; ++c) {
        const int chunk = tid + c * 256;
        const int n = chunk >> 2, q = chunk & 3;
        *(u32x4*)&nxt[n * 48 + q * 8] = pre[c];
      }
    }
    __syncthreads();
  }
}

// ---------------- Edge kernel: fused 3-layer MLP + scatter-add ----------------
__global__ __launch_bounds__(256)
void edge_mlp(const float* __restrict__ x, const int* __restrict__ ei,
              const float* __restrict__ ea,
              const unsigned short* __restrict__ w1a,
              const unsigned short* __restrict__ w1b,
              const unsigned short* __restrict__ w1c,
              const float* __restrict__ b1a, const float* __restrict__ b1b,
              const float* __restrict__ b1c,
              float* __restrict__ summed, float* __restrict__ counts) {
  extern __shared__ unsigned short sm[];
  const int HPAD = 264;
  unsigned short* hA  = sm;                            // 128 x 264 bf16
  unsigned short* hB  = sm + 128 * HPAD;               // 128 x 264 bf16
  unsigned short* wt0 = sm + 2 * 128 * HPAD;           // 256 x 48 bf16
  unsigned short* wt1 = wt0 + 256 * 48;                // 256 x 48 bf16
  int* colsL = (int*)(wt1 + 256 * 48);

  const int tid = threadIdx.x;
  const int ebase = blockIdx.x * 128;

  if (tid < 128) {
    const int c = ei[NE + ebase + tid];
    colsL[tid] = c;
    atomicAdd(&counts[c], 1.0f);
  }
  // Gather h0 = [x[row] || edge_attr]
  for (int r = 0; r < 128; ++r) {
    const int e = ebase + r;
    const int src = ei[e];
    const float v = (tid < DN) ? x[src * DN + tid] : ea[e * DN + (tid - DN)];
    hA[r * HPAD + tid] = f32_to_bf16(v);
  }
  __syncthreads();

  const int lane = tid & 31, wv = tid >> 5;
  const int rbase = wv * 16 + (lane >> 4) * 8;   // C/D: VGPR i -> row rbase+i
  const int cl = lane & 15;                      // C/D: column = nt*16 + cl
  v8f acc[16];

  mlp_layer<256, 16>(hA, HPAD, w1a, wt0, wt1, acc);   // L1 -> relu -> hB
#pragma unroll
  for (int nt = 0; nt < 16; ++nt) {
    const float bb = b1a[nt * 16 + cl];
#pragma unroll
    for (int i = 0; i < 8; ++i)
      hB[(rbase + i) * HPAD + nt * 16 + cl] =
          f32_to_bf16(fmaxf(acc[nt][i] + bb, 0.f));
  }
  mlp_layer<256, 16>(hB, HPAD, w1b, wt0, wt1, acc);   // L2 -> relu -> hA
#pragma unroll
  for (int nt = 0; nt < 16; ++nt) {
    const float bb = b1b[nt * 16 + cl];
#pragma unroll
    for (int i = 0; i < 8; ++i)
      hA[(rbase + i) * HPAD + nt * 16 + cl] =
          f32_to_bf16(fmaxf(acc[nt][i] + bb, 0.f));
  }
  mlp_layer<256, 16>(hA, HPAD, w1c, wt0, wt1, acc);   // L3 -> atomic scatter
#pragma unroll
  for (int nt = 0; nt < 16; ++nt) {
    const float bb = b1c[nt * 16 + cl];
#pragma unroll
    for (int i = 0; i < 8; ++i) {
      const int dest = colsL[rbase + i];
      atomicAdd(&summed[dest * HID + nt * 16 + cl], acc[nt][i] + bb);
    }
  }
}

// ---------------- Node kernel: mean + fused 2-layer MLP ----------------
__global__ __launch_bounds__(256)
void node_mlp(const float* __restrict__ x, const float* __restrict__ summed,
              const float* __restrict__ counts,
              const unsigned short* __restrict__ w2a,
              const unsigned short* __restrict__ w2b,
              const float* __restrict__ b2a, const float* __restrict__ b2b,
              float* __restrict__ out) {
  extern __shared__ unsigned short sm[];
  const int ZPAD = 392, HPAD = 264;
  unsigned short* hZ  = sm;                                // 128 x 392 bf16
  unsigned short* hM  = sm + 128 * ZPAD;                   // 128 x 264 bf16
  unsigned short* wt0 = sm + 128 * ZPAD + 128 * HPAD;      // 256 x 48 bf16
  unsigned short* wt1 = wt0 + 256 * 48;                    // 256 x 48 bf16

  const int tid = threadIdx.x;
  const int nbase = blockIdx.x * 128;

  for (int r = 0; r < 128; ++r) {
    const int node = nbase + r;
    const bool ok = node < NN;
    const float inv = ok ? 1.0f / fmaxf(counts[node], 1.0f) : 0.f;
    for (int c = tid; c < DN + HID; c += 256) {
      float v = 0.f;
      if (ok) v = (c < DN) ? x[node * DN + c] : summed[node * HID + (c - DN)] * inv;
      hZ[r * ZPAD + c] = f32_to_bf16(v);
    }
  }
  __syncthreads();

  const int lane = tid & 31, wv = tid >> 5;
  const int rbase = wv * 16 + (lane >> 4) * 8;
  const int cl = lane & 15;
  v8f acc[16];

  mlp_layer<DN + HID, 16>(hZ, ZPAD, w2a, wt0, wt1, acc);   // 384 -> 256, relu
#pragma unroll
  for (int nt = 0; nt < 16; ++nt) {
    const float bb = b2a[nt * 16 + cl];
#pragma unroll
    for (int i = 0; i < 8; ++i)
      hM[(rbase + i) * HPAD + nt * 16 + cl] =
          f32_to_bf16(fmaxf(acc[nt][i] + bb, 0.f));
  }
  mlp_layer<HID, 8>(hM, HPAD, w2b, wt0, wt1, acc);         // 256 -> 128
#pragma unroll
  for (int nt = 0; nt < 8; ++nt) {
    const float bb = b2b[nt * 16 + cl];
#pragma unroll
    for (int i = 0; i < 8; ++i) {
      const int node = nbase + rbase + i;
      if (node < NN) out[node * DOUT + nt * 16 + cl] = acc[nt][i] + bb;
    }
  }
}

// ---------------- Utility kernels ----------------
__global__ void zero_f32(float* __restrict__ p, int n) {
  for (int i = blockIdx.x * blockDim.x + threadIdx.x; i < n;
       i += gridDim.x * blockDim.x)
    p[i] = 0.f;
}
// Convert f32 [K][N] -> bf16 transposed [N][K]
__global__ void cvt_t_bf16(const float* __restrict__ s,
                           unsigned short* __restrict__ d, int K, int N) {
  const int i = blockIdx.x * blockDim.x + threadIdx.x;
  if (i < K * N) {
    const int n = i / K, k = i % K;
    d[i] = f32_to_bf16(s[k * N + n]);
  }
}

extern "C" void kernel_launch(void* const* d_in, const int* in_sizes, int n_in,
                              void* d_out, int out_size, void* d_ws, size_t ws_size,
                              hipStream_t stream) {
  (void)in_sizes; (void)n_in; (void)out_size; (void)ws_size;
  const float* x   = (const float*)d_in[0];
  const int*   ei  = (const int*)d_in[1];
  const float* ea  = (const float*)d_in[2];
  const float* w1a = (const float*)d_in[5];
  const float* b1a = (const float*)d_in[6];
  const float* w1b = (const float*)d_in[7];
  const float* b1b = (const float*)d_in[8];
  const float* w1c = (const float*)d_in[9];
  const float* b1c = (const float*)d_in[10];
  const float* w2a = (const float*)d_in[11];
  const float* b2a = (const float*)d_in[12];
  const float* w2b = (const float*)d_in[13];
  const float* b2b = (const float*)d_in[14];
  float* out = (float*)d_out;

  // Workspace carve-up
  float* summed = (float*)d_ws;                       // NN*HID f32
  float* counts = summed + (size_t)NN * HID;          // NN f32
  unsigned short* w1a_t = (unsigned short*)(counts + NN);  // transposed bf16
  unsigned short* w1b_t = w1a_t + 256 * 256;
  unsigned short* w1c_t = w1b_t + 256 * 256;
  unsigned short* w2a_t = w1c_t + 256 * 256;               // [256][384]
  unsigned short* w2b_t = w2a_t + 384 * 256;               // [128][256]

  zero_f32<<<1024, 256, 0, stream>>>(summed, NN * HID + NN);
  cvt_t_bf16<<<(256 * 256 + 255) / 256, 256, 0, stream>>>(w1a, w1a_t, 256, 256);
  cvt_t_bf16<<<(256 * 256 + 255) / 256, 256, 0, stream>>>(w1b, w1b_t, 256, 256);
  cvt_t_bf16<<<(256 * 256 + 255) / 256, 256, 0, stream>>>(w1c, w1c_t, 256, 256);
  cvt_t_bf16<<<(384 * 256 + 255) / 256, 256, 0, stream>>>(w2a, w2a_t, 384, 256);
  cvt_t_bf16<<<(256 * 128 + 255) / 256, 256, 0, stream>>>(w2b, w2b_t, 256, 128);

  const size_t edgeLds =
      (size_t)(2 * 128 * 264 + 2 * 256 * 48) * 2 + 128 * 4;   // ~185 KB
  edge_mlp<<<NE / 128, 256, edgeLds, stream>>>(
      x, ei, ea, w1a_t, w1b_t, w1c_t, b1a, b1b, b1c, summed, counts);

  const size_t nodeLds =
      (size_t)(128 * 392 + 128 * 264 + 2 * 256 * 48) * 2;     // ~217 KB
  node_mlp<<<(NN + 127) / 128, 256, nodeLds, stream>>>(
      x, summed, counts, w2a_t, w2b_t, b2a, b2b, out);
}
